// Block_1589137900259
// MI455X (gfx1250) — compile-verified, matrix operand order checked by reading, not run on hardware
//
#include <hip/hip_runtime.h>
#include <math.h>

typedef __bf16 bf16;
typedef __attribute__((ext_vector_type(16))) __bf16 v16bf;
typedef __attribute__((ext_vector_type(8)))  float  v8f;
typedef __attribute__((ext_vector_type(4)))  unsigned u32x4;
typedef __attribute__((ext_vector_type(8)))  int      i32x8;
typedef __attribute__((ext_vector_type(4)))  int      i32x4;

union Frag { uint4 u[2]; v16bf v; };

enum { MODE_BF16 = 0, MODE_VT = 1, MODE_F32RES = 2, MODE_GELU = 3, MODE_SCORES = 5 };

#if defined(__HIP_DEVICE_COMPILE__) && defined(__has_builtin)
#  if __has_builtin(__builtin_amdgcn_tensor_load_to_lds) && __has_builtin(__builtin_amdgcn_s_wait_tensorcnt)
#    define HAVE_TDM 1
#  endif
#endif
#ifndef HAVE_TDM
#  define HAVE_TDM 0
#endif

#define KC 128   // K-chunk staged in LDS (divides 768/1024/3072/9216)

__device__ __forceinline__ bf16 f2bf(float f) {
    union { float f; unsigned u; } cv; cv.f = f;
    unsigned u = cv.u + 0x7fffu + ((cv.u >> 16) & 1u);   // round-to-nearest-even
    union { unsigned short s; bf16 b; } cv2; cv2.s = (unsigned short)(u >> 16);
    return cv2.b;
}

// ---------------------------------------------------------------------------
// bf16 WMMA GEMM: C[M,N] = A[M,K] * B[K,N], B given as BT rows (BT[n][k]
// contiguous in k). Block = 8 waves = 256(M) x 64(N); wave -> 32x64 tile.
// B panel (64 x KC bf16) staged in LDS, double-buffered via the Tensor Data
// Mover (tensor_load_to_lds + s_wait_tensorcnt) when available.
// ---------------------------------------------------------------------------
__global__ __launch_bounds__(256) void gemm_kernel(
    const bf16* __restrict__ A, int lda, long long strideA,
    const bf16* __restrict__ BT, int ldb, long long strideB,
    const float* __restrict__ bias, const float* __restrict__ extra,
    void* __restrict__ out, int ldo, long long strideO,
    int K, float scale, int mode)
{
    __shared__ __align__(16) bf16 sB[2][64 * KC];   // 2 x 16 KB

    const int lane = threadIdx.x & 31;
    const int wave = threadIdx.x >> 5;
    const int m0   = blockIdx.y * 256 + wave * 32;
    const int n0   = blockIdx.x * 64;
    const int z    = blockIdx.z;
    const int row  = lane & 15;
    const int kHi  = (lane >> 4) << 3;       // 0 or 8 (16-bit A/B lane layout)

    const bf16* aRow0  = A  + (long long)z * strideA + (long long)(m0 + row) * lda + kHi;
    const bf16* aRow1  = aRow0 + 16LL * lda;
    const bf16* bPanel = BT + (long long)z * strideB + (long long)n0 * ldb;

    v8f acc[2][4];
#pragma unroll
    for (int h = 0; h < 2; ++h)
#pragma unroll
        for (int t = 0; t < 4; ++t)
#pragma unroll
            for (int r = 0; r < 8; ++r) acc[h][t][r] = 0.f;

    const int nchunks = K / KC;

#if HAVE_TDM
    auto issueTdm = [&](int c, int buf) {
        // D# group0: count=1 | lds_addr | global_addr(57b) | type=2
        unsigned long long ga = (unsigned long long)(size_t)bPanel
                              + (unsigned long long)c * (KC * 2);
        unsigned ldsOff = (unsigned)(size_t)(&sB[buf][0]);
        u32x4 g0;
        g0[0] = 1u;                                       // count=1, user mode
        g0[1] = ldsOff;                                   // lds_addr (bytes)
        g0[2] = (unsigned)(ga & 0xffffffffu);             // global_addr lo
        g0[3] = (unsigned)((ga >> 32) & 0x01ffffffu) | (2u << 30);  // hi | type=2
        // D# group1: mask=0, data_size=1(2B), dims/strides (2-D tile)
        i32x8 g1;
        g1[0] = 0x00010000;                               // data_size = 2 bytes
        g1[1] = (int)((unsigned)(KC & 0xffff) << 16);     // tensor_dim0 [15:0]
        g1[2] = (int)((KC >> 16) | (64u << 16));          // tdim0 hi | tensor_dim1 lo
        g1[3] = (int)((unsigned)KC << 16);                // tdim1 hi=0 | tile_dim0=KC
        g1[4] = 64;                                       // tile_dim1=64, tile_dim2=0
        g1[5] = ldb;                                      // tensor_dim0_stride lo
        g1[6] = 0;                                        // stride hi | dim1_stride lo
        g1[7] = 0;
        i32x4 gz = { 0, 0, 0, 0 };                        // groups 2/3 unused (2-D)
#  if __clang_major__ >= 23
        i32x8 gz8 = { 0, 0, 0, 0, 0, 0, 0, 0 };
        __builtin_amdgcn_tensor_load_to_lds(g0, g1, gz, gz, gz8, 0);
#  else
        __builtin_amdgcn_tensor_load_to_lds(g0, g1, gz, gz, 0);
#  endif
    };
    if (wave == 0) issueTdm(0, 0);
#endif

    for (int c = 0; c < nchunks; ++c) {
        const int cur = c & 1;
#if HAVE_TDM
        __syncthreads();                       // buf cur^1 free for overwrite
        if (wave == 0) {
            if (c + 1 < nchunks) {
                issueTdm(c + 1, cur ^ 1);
                __builtin_amdgcn_s_wait_tensorcnt(1);   // chunk c complete
            } else {
                __builtin_amdgcn_s_wait_tensorcnt(0);
            }
        }
        __syncthreads();                       // chunk c visible to all waves
#else
        __syncthreads();
        // cooperative global -> LDS copy of chunk c (fallback path)
        for (int i = threadIdx.x; i < 64 * KC / 8; i += 256) {
            int r = i >> 4, kk = (i & 15) << 3;          // KC/8 == 16
            *(uint4*)&sB[cur][r * KC + kk] =
                *(const uint4*)&bPanel[(long long)r * ldb + c * KC + kk];
        }
        __syncthreads();
#endif
#pragma unroll
        for (int ks = 0; ks < KC / 32; ++ks) {
            const int kk = ks * 32 + kHi;
            Frag a0, a1;
            const bf16* ap0 = aRow0 + c * KC + ks * 32;
            const bf16* ap1 = aRow1 + c * KC + ks * 32;
            a0.u[0] = *(const uint4*)(ap0);
            a0.u[1] = *(const uint4*)(ap0 + 16);
            a1.u[0] = *(const uint4*)(ap1);
            a1.u[1] = *(const uint4*)(ap1 + 16);
#pragma unroll
            for (int t = 0; t < 4; ++t) {
                Frag b;
                const bf16* bt = &sB[cur][(t * 16 + row) * KC + kk];
                b.u[0] = *(const uint4*)(bt);
                b.u[1] = *(const uint4*)(bt + 16);
                acc[0][t] = __builtin_amdgcn_wmma_f32_16x16x32_bf16(
                    false, a0.v, false, b.v, (short)0, acc[0][t], false, false);
                acc[1][t] = __builtin_amdgcn_wmma_f32_16x16x32_bf16(
                    false, a1.v, false, b.v, (short)0, acc[1][t], false, false);
            }
        }
    }

    // C/D layout: VGPR r -> M = r (lanes 0-15) or r+8 (lanes 16-31); N = lane&15
    const int colBase = n0 + row;
#pragma unroll
    for (int h = 0; h < 2; ++h) {
        const int mBase = m0 + h * 16 + ((lane >> 4) << 3);
#pragma unroll
        for (int t = 0; t < 4; ++t) {
            const int col = colBase + t * 16;
            const float bval = bias ? bias[col] : 0.f;
#pragma unroll
            for (int r = 0; r < 8; ++r) {
                const int m = mBase + r;
                float v = acc[h][t][r];
                if (mode == MODE_BF16) {
                    ((bf16*)out)[z * strideO + (long long)m * ldo + col] = f2bf(v + bval);
                } else if (mode == MODE_VT) {
                    // store v transposed: out[nb][e][s], nb = m/1024, s = m%1024
                    int nb = m >> 10, s = m & 1023;
                    ((bf16*)out)[(long long)nb * (768 * 1024) + (long long)col * 1024 + s] =
                        f2bf(v + bval);
                } else if (mode == MODE_F32RES) {
                    long long idx = (long long)m * ldo + col;
                    ((float*)out)[z * strideO + idx] = v + bval + extra[idx];
                } else if (mode == MODE_GELU) {
                    float g = v + bval;
                    g = 0.5f * g * (1.0f + erff(g * 0.70710678118654752f));
                    ((bf16*)out)[z * strideO + (long long)m * ldo + col] = f2bf(g);
                } else { // MODE_SCORES: scale + causal mask, f32
                    float sv = v * scale;
                    if (col > m) sv = -__builtin_inff();
                    ((float*)out)[z * strideO + (long long)m * ldo + col] = sv;
                }
            }
        }
    }
}

// ---------------------------------------------------------------------------
// LayerNorm over E=768, one block per row; writes f32 and bf16 copies.
// ---------------------------------------------------------------------------
__global__ __launch_bounds__(256) void ln_kernel(
    const float* __restrict__ in, const float* __restrict__ g,
    const float* __restrict__ b, float* __restrict__ of, bf16* __restrict__ ob)
{
    constexpr int E = 768;
    const int tid = threadIdx.x;
    const long long base = (long long)blockIdx.x * E;
    __shared__ float red[256];
    float l0 = in[base + tid], l1 = in[base + tid + 256], l2 = in[base + tid + 512];
    red[tid] = l0 + l1 + l2;
    __syncthreads();
    for (int o = 128; o > 0; o >>= 1) { if (tid < o) red[tid] += red[tid + o]; __syncthreads(); }
    float mean = red[0] * (1.0f / E);
    __syncthreads();
    float d0 = l0 - mean, d1 = l1 - mean, d2 = l2 - mean;
    red[tid] = d0 * d0 + d1 * d1 + d2 * d2;
    __syncthreads();
    for (int o = 128; o > 0; o >>= 1) { if (tid < o) red[tid] += red[tid + o]; __syncthreads(); }
    float rstd = rsqrtf(red[0] * (1.0f / E) + 1e-5f);
    float d[3] = { d0, d1, d2 };
#pragma unroll
    for (int i = 0; i < 3; ++i) {
        int c = tid + 256 * i;
        float o = d[i] * rstd * g[c] + b[c];
        of[base + c] = o;
        ob[base + c] = f2bf(o);
    }
}

// ---------------------------------------------------------------------------
// Row softmax over 1024 fp32 scores (-inf masked) -> bf16 probs.
// ---------------------------------------------------------------------------
__global__ __launch_bounds__(256) void softmax_kernel(
    const float* __restrict__ scs, bf16* __restrict__ attn)
{
    const int tid = threadIdx.x;
    const long long base = (long long)blockIdx.x * 1024;
    __shared__ float red[256];
    float l0 = scs[base + tid],       l1 = scs[base + tid + 256];
    float l2 = scs[base + tid + 512], l3 = scs[base + tid + 768];
    red[tid] = fmaxf(fmaxf(l0, l1), fmaxf(l2, l3));
    __syncthreads();
    for (int o = 128; o > 0; o >>= 1) { if (tid < o) red[tid] = fmaxf(red[tid], red[tid + o]); __syncthreads(); }
    float mx = red[0];
    __syncthreads();
    float e0 = __expf(l0 - mx), e1 = __expf(l1 - mx);
    float e2 = __expf(l2 - mx), e3 = __expf(l3 - mx);
    red[tid] = e0 + e1 + e2 + e3;
    __syncthreads();
    for (int o = 128; o > 0; o >>= 1) { if (tid < o) red[tid] += red[tid + o]; __syncthreads(); }
    float inv = 1.0f / red[0];
    attn[base + tid]       = f2bf(e0 * inv);
    attn[base + tid + 256] = f2bf(e1 * inv);
    attn[base + tid + 512] = f2bf(e2 * inv);
    attn[base + tid + 768] = f2bf(e3 * inv);
}

// ---------------------------------------------------------------------------
// Transpose + convert: f32 [R][C] -> bf16 [C][R]. 32x32 LDS tiles.
// ---------------------------------------------------------------------------
__global__ __launch_bounds__(256) void transcvt_kernel(
    const float* __restrict__ in, bf16* __restrict__ out,
    int R, int C, long long strideIn, long long strideOut)
{
    __shared__ float tile[32][33];
    in  += (long long)blockIdx.z * strideIn;
    out += (long long)blockIdx.z * strideOut;
    const int c0 = blockIdx.x * 32, r0 = blockIdx.y * 32;
    const int tx = threadIdx.x & 31, ty = threadIdx.x >> 5;  // 32 x 8
#pragma unroll
    for (int i = 0; i < 32; i += 8)
        tile[ty + i][tx] = in[(long long)(r0 + ty + i) * C + c0 + tx];
    __syncthreads();
#pragma unroll
    for (int i = 0; i < 32; i += 8)
        out[(long long)(c0 + ty + i) * R + r0 + tx] = f2bf(tile[tx][ty + i]);
}

// ---------------------------------------------------------------------------
extern "C" void kernel_launch(void* const* d_in, const int* in_sizes, int n_in,
                              void* d_out, int out_size, void* d_ws, size_t ws_size,
                              hipStream_t stream) {
    (void)in_sizes; (void)n_in; (void)out_size;
    const float* inputs = (const float*)d_in[0];
    const float* g1  = (const float*)d_in[1];
    const float* b1  = (const float*)d_in[2];
    const float* Wq  = (const float*)d_in[3];
    const float* bq  = (const float*)d_in[4];
    const float* Wk  = (const float*)d_in[5];
    const float* bk  = (const float*)d_in[6];
    const float* Wv  = (const float*)d_in[7];
    const float* bv  = (const float*)d_in[8];
    const float* Wc  = (const float*)d_in[9];
    const float* bc  = (const float*)d_in[10];
    const float* g2  = (const float*)d_in[11];
    const float* b2  = (const float*)d_in[12];
    const float* Wfc = (const float*)d_in[13];
    const float* bfc = (const float*)d_in[14];
    const float* Wp  = (const float*)d_in[15];
    const float* bp  = (const float*)d_in[16];

    constexpr int NB = 4, S = 1024, E = 768, H = 12;
    constexpr int M  = NB * S;   // 4096
    constexpr int HE = H * E;    // 9216
    constexpr int FF = 4 * E;    // 3072

    char* base = (char*)d_ws;
    size_t off = 0;
    auto alloc = [&](size_t bytes) -> void* {
        void* p = base + off;
        off = (off + bytes + 255) & ~(size_t)255;
        return p;
    };
    float* x_f  = (float*)alloc((size_t)M * E * 4);
    bf16*  x_b  = (bf16*) alloc((size_t)M * E * 2);
    bf16*  WqT  = (bf16*) alloc((size_t)H * E * E * 2);
    bf16*  WkT  = (bf16*) alloc((size_t)H * E * E * 2);
    bf16*  WvT  = (bf16*) alloc((size_t)H * E * E * 2);
    bf16*  WcT  = (bf16*) alloc((size_t)HE * E * 2);
    bf16*  WfcT = (bf16*) alloc((size_t)E * FF * 2);
    bf16*  WpT  = (bf16*) alloc((size_t)FF * E * 2);
    bf16*  qh   = (bf16*) alloc((size_t)M * E * 2);
    bf16*  kh   = (bf16*) alloc((size_t)M * E * 2);
    bf16*  vTh  = (bf16*) alloc((size_t)M * E * 2);
    float* scs  = (float*)alloc((size_t)NB * S * S * 4);
    bf16*  attn = (bf16*) alloc((size_t)NB * S * S * 2);
    bf16*  catb = (bf16*) alloc((size_t)M * HE * 2);
    float* ybuf = (float*)alloc((size_t)M * E * 4);
    float* y2f  = (float*)alloc((size_t)M * E * 4);
    bf16*  y2b  = (bf16*) alloc((size_t)M * E * 2);
    bf16*  hb   = (bf16*) alloc((size_t)M * FF * 2);
    if (off > ws_size) return;   // workspace too small; nothing safe to do

    dim3 blk(256);

    // LN1 -> x (f32 + bf16)
    ln_kernel<<<M, blk, 0, stream>>>(inputs, g1, b1, x_f, x_b);

    // weights -> bf16, transposed to [out][in]
    transcvt_kernel<<<dim3(E/32, E/32, H), blk, 0, stream>>>(Wq, WqT, E, E, (long long)E*E, (long long)E*E);
    transcvt_kernel<<<dim3(E/32, E/32, H), blk, 0, stream>>>(Wk, WkT, E, E, (long long)E*E, (long long)E*E);
    transcvt_kernel<<<dim3(E/32, E/32, H), blk, 0, stream>>>(Wv, WvT, E, E, (long long)E*E, (long long)E*E);
    transcvt_kernel<<<dim3(E/32, HE/32, 1), blk, 0, stream>>>(Wc,  WcT,  HE, E,  0, 0);
    transcvt_kernel<<<dim3(FF/32, E/32, 1), blk, 0, stream>>>(Wfc, WfcT, E,  FF, 0, 0);
    transcvt_kernel<<<dim3(E/32, FF/32, 1), blk, 0, stream>>>(Wp,  WpT,  FF, E,  0, 0);

    const float sc = 0.03608439182435161f;   // 1/sqrt(768)
    for (int h = 0; h < H; ++h) {
        const bf16* WqTh = WqT + (size_t)h * E * E;
        const bf16* WkTh = WkT + (size_t)h * E * E;
        const bf16* WvTh = WvT + (size_t)h * E * E;
        // q = x @ Wq[h] + bq[h]   (bf16 out [M][E])
        gemm_kernel<<<dim3(E/64, M/256, 1), blk, 0, stream>>>(
            x_b, E, 0, WqTh, E, 0, bq + h*E, nullptr, qh, E, 0, E, 1.f, MODE_BF16);
        // k
        gemm_kernel<<<dim3(E/64, M/256, 1), blk, 0, stream>>>(
            x_b, E, 0, WkTh, E, 0, bk + h*E, nullptr, kh, E, 0, E, 1.f, MODE_BF16);
        // v, stored transposed [nb][e][s] for the PV GEMM B-side
        gemm_kernel<<<dim3(E/64, M/256, 1), blk, 0, stream>>>(
            x_b, E, 0, WvTh, E, 0, bv + h*E, nullptr, vTh, 0, 0, E, 1.f, MODE_VT);
        // scores = q k^T / sqrt(E), causal mask; batch over nb
        gemm_kernel<<<dim3(S/64, S/256, NB), blk, 0, stream>>>(
            qh, E, (long long)S*E, kh, E, (long long)S*E, nullptr, nullptr,
            scs, S, (long long)S*S, E, sc, MODE_SCORES);
        // softmax rows -> bf16 probs
        softmax_kernel<<<NB*S, blk, 0, stream>>>(scs, attn);
        // heads = attn @ v -> cat[:, :, h*E : (h+1)*E]
        gemm_kernel<<<dim3(E/64, S/256, NB), blk, 0, stream>>>(
            attn, S, (long long)S*S, vTh, S, (long long)E*S, nullptr, nullptr,
            catb + h*E, HE, (long long)S*HE, S, 1.f, MODE_BF16);
    }

    // y = cat @ Wc + bc + x
    gemm_kernel<<<dim3(E/64, M/256, 1), blk, 0, stream>>>(
        catb, HE, 0, WcT, HE, 0, bc, x_f, ybuf, E, 0, HE, 1.f, MODE_F32RES);
    // y2 = LN(y)
    ln_kernel<<<M, blk, 0, stream>>>(ybuf, g2, b2, y2f, y2b);
    // h = gelu(y2 @ Wfc + bfc)
    gemm_kernel<<<dim3(FF/64, M/256, 1), blk, 0, stream>>>(
        y2b, E, 0, WfcT, E, 0, bfc, nullptr, hb, FF, 0, E, 1.f, MODE_GELU);
    // out = h @ Wp + bp + y2
    gemm_kernel<<<dim3(E/64, M/256, 1), blk, 0, stream>>>(
        hb, FF, 0, WpT, FF, 0, bp, y2f, (float*)d_out, E, 0, FF, 1.f, MODE_F32RES);
}
